// k1_GNN_sub_multi_h_7842610283384
// MI455X (gfx1250) — compile-verified, matrix operand order checked by reading, not run on hardware
//
#include <hip/hip_runtime.h>
#include <hip/hip_bf16.h>

typedef __attribute__((ext_vector_type(16))) _Float16 v16h;
typedef __attribute__((ext_vector_type(8)))  float    v8f;

#define N_NODES  8192
#define N_EDGES  65536
#define G_GRAPHS 16
#define S_SUB    32
#define NUM_HH   4
#define EA_DIM   5
#define HID      128

// ---------------------------------------------------------------- utilities
__global__ void zero_kernel(float* p, int n) {
  int i = blockIdx.x * blockDim.x + threadIdx.x;
  if (i < n) p[i] = 0.f;
}

__global__ void cvt_x16_kernel(const float* __restrict__ x, _Float16* __restrict__ o, int n) {
  int i = blockIdx.x * blockDim.x + threadIdx.x;
  if (i < n) o[i] = (_Float16)x[i];
}

// --------------------------------------------------- h = relu(ea @ w1 + b1)
__global__ __launch_bounds__(HID) void edge_mlp_kernel(const float* __restrict__ ea,
    const float* __restrict__ w1, const float* __restrict__ b1,
    _Float16* __restrict__ h16) {
  __shared__ float eas[EA_DIM];
  const int e = blockIdx.x;
  if (threadIdx.x < EA_DIM) eas[threadIdx.x] = ea[e * EA_DIM + threadIdx.x];
  __syncthreads();
  const int t = threadIdx.x;
  float s = b1[t];
#pragma unroll
  for (int j = 0; j < EA_DIM; ++j) s += eas[j] * w1[j * HID + t];
  s = fmaxf(s, 0.f);
  h16[(size_t)e * HID + t] = (_Float16)s;
}

// ----- repack w2 (+ b2 as extra K rows) into per-lane WMMA B-fragment order
// W2'[m, o]: m = k*din + i  (m < 128*din comes from w2; m >= 128*din from b2)
// storage: [((chunk*NT + ntile)*32 + lane)*16 + j], j in fragment element order
__global__ void pack_w2_kernel(const float* __restrict__ w2, const float* __restrict__ b2,
    _Float16* __restrict__ w2p, int din, int dout, int logdin, int total) {
  int idx = blockIdx.x * blockDim.x + threadIdx.x;
  if (idx >= total) return;
  const int ktot = HID * din;
  const int nt = dout >> 4;
  int j     = idx & 15;
  int lane  = (idx >> 4) & 31;
  int rest  = idx >> 9;
  int ntile = rest % nt;
  int ch    = rest / nt;
  int hi    = lane >> 4;
  int kj    = (j < 8) ? (hi * 8 + j) : (8 + hi * 8 + j);  // = 16 + hi*8 + (j-8)
  int m     = ch * 32 + kj;
  int o     = ntile * 16 + (lane & 15);
  float v;
  if (m < ktot) {
    int kh = m >> logdin;
    int xi = m & (din - 1);
    v = w2[(size_t)kh * (din * dout) + (size_t)xi * dout + o];
  } else {
    int i2 = m - ktot;
    v = (i2 < din) ? b2[i2 * dout + o] : 0.f;
  }
  w2p[idx] = (_Float16)v;
}

// --------------------------------------------------- fused message GEMM
// msg[e,:] = (h[e] (x) x[src[e]]) @ W2'  via v_wmma_f32_16x16x32_f16.
// Block = 2 edge-groups (64 edges, LDS-shared) x 2 N-halves.
// Each wave: 4 A tiles x NT/2 n-tiles -> every B fragment feeds 4 WMMAs.
template <int DIN, int DOUT, int LOG2DIN>
__global__ __launch_bounds__(128) void msg_wmma_kernel(
    const _Float16* __restrict__ h16, const _Float16* __restrict__ x16,
    const int* __restrict__ src, const int* __restrict__ dst,
    const _Float16* __restrict__ w2p, float* __restrict__ agg) {
  constexpr int NCH   = (HID * DIN) / 32;   // K chunks from h (x) x
  constexpr int EXTRA = (DIN + 31) / 32;    // K chunks carrying b2 (A = x)
  constexpr int NT    = DOUT / 16;          // N tiles total
  constexpr int NTW   = NT / 2;             // N tiles per wave (half)

  const int wave = threadIdx.x >> 5;
  const int lane = threadIdx.x & 31;
  const int grp  = wave >> 1;               // edge group within block
  const int nh   = wave & 1;                // N-half owned by this wave
  const int e0   = (blockIdx.x * 2 + grp) * 64;
  const int row  = lane & 15;
  const int hi   = lane >> 4;

  __shared__ __align__(16) _Float16 h_s[2][64][HID];
  __shared__ __align__(16) _Float16 x_s[2][64][DIN];
  __shared__ int dst_s[2][64];

  // the wave-pair of a group stages cooperatively (64 lanes, stride 64)
  {
    const int t0 = nh * 32 + lane;
    dst_s[grp][t0] = dst[e0 + t0];
    for (int t = t0; t < 64 * (HID / 8); t += 64) {
      int r = t >> 4, c = t & 15;
      ((uint4*)(&h_s[grp][r][0]))[c] =
          ((const uint4*)(h16 + (size_t)(e0 + r) * HID))[c];
    }
    for (int t = t0; t < 64 * (DIN / 8); t += 64) {
      int r = t / (DIN / 8), c = t % (DIN / 8);
      int s = src[e0 + r];
      ((uint4*)(&x_s[grp][r][0]))[c] =
          ((const uint4*)(x16 + (size_t)s * DIN))[c];
    }
  }
  __syncthreads();

  v8f acc[4][NTW] = {};

  for (int ch = 0; ch < NCH; ++ch) {
    __builtin_prefetch(w2p + ((size_t)(ch + 1) * NT * 32) * 16, 0, 1);
    v16h a[4];
#pragma unroll
    for (int j = 0; j < 16; ++j) {
      const int kj = (j < 8) ? (hi * 8 + j) : (8 + hi * 8 + j);
      const int m  = ch * 32 + kj;
      const int kh = m >> LOG2DIN;
      const int xi = m & (DIN - 1);
#pragma unroll
      for (int et = 0; et < 4; ++et) {
        const int r = et * 16 + row;
        a[et][j] = h_s[grp][r][kh] * x_s[grp][r][xi];
      }
    }
#pragma unroll
    for (int n = 0; n < NTW; ++n) {
      const int ntg = nh * NTW + n;
      const v16h b = *(const v16h*)(w2p + ((size_t)(ch * NT + ntg) * 32 + lane) * 16);
#pragma unroll
      for (int et = 0; et < 4; ++et)
        acc[et][n] = __builtin_amdgcn_wmma_f32_16x16x32_f16(
            false, a[et], false, b, (short)0, acc[et][n], false, false);
    }
  }
  // b2 contribution: A rows are x[src] itself
#pragma unroll
  for (int ce = 0; ce < EXTRA; ++ce) {
    const int ch = NCH + ce;
    v16h a[4];
#pragma unroll
    for (int j = 0; j < 16; ++j) {
      const int kj = (j < 8) ? (hi * 8 + j) : (8 + hi * 8 + j);
      const int m2 = ce * 32 + kj;
#pragma unroll
      for (int et = 0; et < 4; ++et)
        a[et][j] = (m2 < DIN) ? x_s[grp][et * 16 + row][m2] : (_Float16)0.f;
    }
#pragma unroll
    for (int n = 0; n < NTW; ++n) {
      const int ntg = nh * NTW + n;
      const v16h b = *(const v16h*)(w2p + ((size_t)(ch * NT + ntg) * 32 + lane) * 16);
#pragma unroll
      for (int et = 0; et < 4; ++et)
        acc[et][n] = __builtin_amdgcn_wmma_f32_16x16x32_f16(
            false, a[et], false, b, (short)0, acc[et][n], false, false);
    }
  }

  // C/D layout: VGPR r -> M = r + 8*hi ; N = ntg*16 + (lane&15)
#pragma unroll
  for (int et = 0; et < 4; ++et) {
#pragma unroll
    for (int n = 0; n < NTW; ++n) {
      const int ntg = nh * NTW + n;
#pragma unroll
      for (int r = 0; r < 8; ++r) {
        const int er = et * 16 + hi * 8 + r;
        const int d  = dst_s[grp][er];
        atomicAdd(&agg[(size_t)d * DOUT + ntg * 16 + row], acc[et][n][r]);
      }
    }
  }
}

// --------------------------- x' = elu(agg + x @ root + bias); also emit f16
__global__ void node_update_kernel(const float* __restrict__ agg, const float* __restrict__ xin,
    const float* __restrict__ root, const float* __restrict__ bias,
    float* __restrict__ xout, _Float16* __restrict__ x16out,
    int din, int dout, int total) {
  int idx = blockIdx.x * blockDim.x + threadIdx.x;
  if (idx >= total) return;
  int n = idx / dout, c = idx % dout;
  float s = agg[idx] + bias[c];
  for (int i = 0; i < din; ++i) s += xin[(size_t)n * din + i] * root[i * dout + c];
  float y = (s > 0.f) ? s : (expf(s) - 1.f);
  xout[idx] = y;
  x16out[idx] = (_Float16)y;
}

// ------------------------------------------------------------- pooling + FC
__global__ void pool_sum_kernel(const float* __restrict__ xf, const int* __restrict__ n2s,
    float* __restrict__ sums, float* __restrict__ cnts) {
  int idx = blockIdx.x * blockDim.x + threadIdx.x;
  if (idx >= N_NODES * 64) return;
  int n = idx >> 6, c = idx & 63;
  int s = n2s[n];
  atomicAdd(&sums[s * 64 + c], xf[idx]);
  if (c == 0) atomicAdd(&cnts[s], 1.f);
}

__global__ void graph_feat_kernel(const float* __restrict__ sums, const float* __restrict__ cnts,
    float* __restrict__ xg) {
  int idx = blockIdx.x * blockDim.x + threadIdx.x;  // G*256
  if (idx >= G_GRAPHS * 256) return;
  int g = idx >> 8, f = idx & 255;
  int hh = f >> 6, c = f & 63;
  const int inc = S_SUB / NUM_HH;  // 8
  float acc = 0.f;
  for (int j = 0; j < inc; ++j) {
    int sg = g * S_SUB + hh * inc + j;
    acc += sums[sg * 64 + c] / cnts[sg];
  }
  xg[idx] = acc / (float)inc;
}

__global__ __launch_bounds__(256) void fc_head_kernel(const float* __restrict__ xg,
    const float* __restrict__ w1, const float* __restrict__ b1,
    const float* __restrict__ w2, const float* __restrict__ b2,
    const float* __restrict__ w3, const float* __restrict__ b3,
    float* __restrict__ out) {
  __shared__ float xs[256], h1[128], h2[64];
  const int g = blockIdx.x, t = threadIdx.x;
  xs[t] = xg[g * 256 + t];
  __syncthreads();
  if (t < 128) {
    float s = b1[t];
    for (int i = 0; i < 256; ++i) s += xs[i] * w1[i * 128 + t];
    h1[t] = (s > 0.f) ? s : (expf(s) - 1.f);
  }
  __syncthreads();
  if (t < 64) {
    float s = b2[t];
    for (int i = 0; i < 128; ++i) s += h1[i] * w2[i * 64 + t];
    h2[t] = (s > 0.f) ? s : (expf(s) - 1.f);
  }
  __syncthreads();
  if (t == 0) {
    float s = b3[0];
    for (int i = 0; i < 64; ++i) s += h2[i] * w3[i];
    out[g] = s;
  }
}

// ------------------------------------------------------------------- driver
extern "C" void kernel_launch(void* const* d_in, const int* in_sizes, int n_in,
                              void* d_out, int out_size, void* d_ws, size_t ws_size,
                              hipStream_t stream) {
  (void)in_sizes; (void)n_in; (void)out_size; (void)ws_size;
  const float* x_in = (const float*)d_in[0];
  const int*   eidx = (const int*)d_in[1];
  const float* ea   = (const float*)d_in[2];
  const int*   n2s  = (const int*)d_in[3];
  const float* P[24];
  for (int i = 0; i < 24; ++i) P[i] = (const float*)d_in[5 + i];
  const int* src = eidx;
  const int* dst = eidx + N_EDGES;

  // workspace carve
  char* wp = (char*)d_ws;
  auto carve = [&](size_t bytes) { void* p = wp; wp += (bytes + 255) & ~(size_t)255; return p; };
  _Float16* h16   = (_Float16*)carve((size_t)N_EDGES * HID * 2);        // 16 MB
  _Float16* w2p   = (_Float16*)carve((size_t)258 * 32 * 64 * 2);        // ~1 MB (max layer)
  float*    xA    = (float*)carve((size_t)N_NODES * 64 * 4);
  float*    xB    = (float*)carve((size_t)N_NODES * 64 * 4);
  _Float16* x16a  = (_Float16*)carve((size_t)N_NODES * 64 * 2);
  _Float16* x16b  = (_Float16*)carve((size_t)N_NODES * 64 * 2);
  float*    agg   = (float*)carve((size_t)N_NODES * 64 * 4);
  float*    xsum  = (float*)carve((size_t)G_GRAPHS * S_SUB * 64 * 4);
  float*    xcnt  = (float*)carve((size_t)G_GRAPHS * S_SUB * 4);
  float*    xg    = (float*)carve((size_t)G_GRAPHS * 256 * 4);

  cvt_x16_kernel<<<(N_NODES * 16 + 255) / 256, 256, 0, stream>>>(x_in, x16a, N_NODES * 16);

  // ---------------- layer 0: din=16, dout=32
  {
    const int din = 16, dout = 32, logd = 4;
    const int ncht = 4 * din + (din + 31) / 32;
    const int ptot = ncht * 32 * dout;
    edge_mlp_kernel<<<N_EDGES, HID, 0, stream>>>(ea, P[0], P[1], h16);
    pack_w2_kernel<<<(ptot + 255) / 256, 256, 0, stream>>>(P[2], P[3], w2p, din, dout, logd, ptot);
    zero_kernel<<<(N_NODES * dout + 255) / 256, 256, 0, stream>>>(agg, N_NODES * dout);
    msg_wmma_kernel<16, 32, 4><<<N_EDGES / 128, 128, 0, stream>>>(h16, x16a, src, dst, w2p, agg);
    node_update_kernel<<<(N_NODES * dout + 255) / 256, 256, 0, stream>>>(
        agg, x_in, P[4], P[5], xB, x16b, din, dout, N_NODES * dout);
  }
  // ---------------- layer 1: din=32, dout=64
  {
    const int din = 32, dout = 64, logd = 5;
    const int ncht = 4 * din + (din + 31) / 32;
    const int ptot = ncht * 32 * dout;
    edge_mlp_kernel<<<N_EDGES, HID, 0, stream>>>(ea, P[6], P[7], h16);
    pack_w2_kernel<<<(ptot + 255) / 256, 256, 0, stream>>>(P[8], P[9], w2p, din, dout, logd, ptot);
    zero_kernel<<<(N_NODES * dout + 255) / 256, 256, 0, stream>>>(agg, N_NODES * dout);
    msg_wmma_kernel<32, 64, 5><<<N_EDGES / 128, 128, 0, stream>>>(h16, x16b, src, dst, w2p, agg);
    node_update_kernel<<<(N_NODES * dout + 255) / 256, 256, 0, stream>>>(
        agg, xB, P[10], P[11], xA, x16a, din, dout, N_NODES * dout);
  }
  // ---------------- layer 2: din=64, dout=64
  {
    const int din = 64, dout = 64, logd = 6;
    const int ncht = 4 * din + (din + 31) / 32;
    const int ptot = ncht * 32 * dout;
    edge_mlp_kernel<<<N_EDGES, HID, 0, stream>>>(ea, P[12], P[13], h16);
    pack_w2_kernel<<<(ptot + 255) / 256, 256, 0, stream>>>(P[14], P[15], w2p, din, dout, logd, ptot);
    zero_kernel<<<(N_NODES * dout + 255) / 256, 256, 0, stream>>>(agg, N_NODES * dout);
    msg_wmma_kernel<64, 64, 6><<<N_EDGES / 128, 128, 0, stream>>>(h16, x16a, src, dst, w2p, agg);
    node_update_kernel<<<(N_NODES * dout + 255) / 256, 256, 0, stream>>>(
        agg, xA, P[16], P[17], xB, x16b, din, dout, N_NODES * dout);
  }

  // ---------------- pooling + head
  zero_kernel<<<(G_GRAPHS * S_SUB * 64 + 255) / 256, 256, 0, stream>>>(xsum, G_GRAPHS * S_SUB * 64);
  zero_kernel<<<(G_GRAPHS * S_SUB + 255) / 256, 256, 0, stream>>>(xcnt, G_GRAPHS * S_SUB);
  pool_sum_kernel<<<(N_NODES * 64 + 255) / 256, 256, 0, stream>>>(xB, n2s, xsum, xcnt);
  graph_feat_kernel<<<(G_GRAPHS * 256 + 255) / 256, 256, 0, stream>>>(xsum, xcnt, xg);
  fc_head_kernel<<<G_GRAPHS, 256, 0, stream>>>(xg, P[18], P[19], P[20], P[21], P[22], P[23],
                                               (float*)d_out);
}
